// Atten_50611894616900
// MI455X (gfx1250) — matrix operand order repaired and checked by reading, not compiled, not run on hardware
//
#include <hip/hip_runtime.h>
#include <hip/hip_bf16.h>
#include <stdint.h>

// Problem constants (match reference)
#define N_POINTS 32768
#define K_NB     16
#define CH       128        // ch_in == HD == 128
#define INNER    512
#define ROWS_PER_WG 128
#define NWG      (N_POINTS / ROWS_PER_WG)   // 256
#define BN_EPS   1e-5f

typedef __attribute__((ext_vector_type(16))) __bf16 v16bf;
typedef __attribute__((ext_vector_type(8)))  float  v8f;

__device__ __forceinline__ uint32_t f2bf_bits(float f) {
    uint32_t u = __builtin_bit_cast(uint32_t, f);
    // round-to-nearest-even bf16 truncation
    return (u + 0x7FFFu + ((u >> 16) & 1u)) >> 16;
}
__device__ __forceinline__ uint32_t pack2bf(float lo, float hi) {
    return f2bf_bits(lo) | (f2bf_bits(hi) << 16);
}

// ---------------------------------------------------------------------------
// Kernel 1: W = Wv @ Wo (f32), stored directly in WMMA B-fragment friendly
// packing: WtP[j][c2] = { bf16 W[2*c2][j], bf16 W[2*c2+1][j] } (pairs along K=c)
// 8192 entries, one per thread. One-time tiny GEMM (16.8 MFLOP).
// ---------------------------------------------------------------------------
__global__ void prep_w_kernel(const float* __restrict__ Wv,
                              const float* __restrict__ Wo,
                              uint32_t* __restrict__ WtP) {
    int tid = blockIdx.x * blockDim.x + threadIdx.x;   // 0..8191
    int j  = tid >> 6;          // output channel 0..127
    int c2 = tid & 63;          // K pair index
    int c0 = c2 * 2;
    float s0 = 0.f, s1 = 0.f;
    for (int i = 0; i < INNER; ++i) {
        float wo = Wo[(size_t)i * CH + j];
        s0 = fmaf(Wv[(size_t)c0 * INNER + i],       wo, s0);
        s1 = fmaf(Wv[(size_t)(c0 + 1) * INNER + i], wo, s1);
    }
    WtP[tid] = pack2bf(s0, s1);
}

// ---------------------------------------------------------------------------
// Kernel 2: fused  hs = sum_k h[n,k,:]  ->  out = hs @ W + bo  (WMMA bf16)
// plus deterministic per-WG BN partial sums.
// 256 threads = 8 wave32; wave w owns rows [16w, 16w+16) of the WG's 128 rows.
// ---------------------------------------------------------------------------
__global__ __launch_bounds__(256) void main_kernel(
    const float*    __restrict__ h,
    const uint32_t* __restrict__ WtP,
    const float*    __restrict__ bo,
    float*          __restrict__ outp,      // [N, CH] pre-BN (== d_out)
    float*          __restrict__ part_sum,  // [NWG, CH]
    float*          __restrict__ part_sq)   // [NWG, CH]
{
    // +1 dword pad per row -> bank-conflict-free fragment gathers (stride 65)
    __shared__ uint32_t sW[128][65];   // B-packed W: [j][c2]
    __shared__ uint32_t sA[128][65];   // A-packed hs: [row][c2]
    __shared__ float    sRS[16][128];  // per (wave,half) channel sums
    __shared__ float    sRQ[16][128];  // per (wave,half) channel sumsq

    const int tid = threadIdx.x;
    const int n0  = blockIdx.x * ROWS_PER_WG;

    // --- stage packed W into LDS (32 KB, coalesced) ---
    for (int i = tid; i < 128 * 64; i += 256)
        sW[i >> 6][i & 63] = WtP[i];

    // --- stream h, reduce over K in registers, pack hs->bf16 into LDS ---
    const int c4    = tid & 31;   // float4 channel group (covers CH=128)
    const int rbase = tid >> 5;   // 0..7
    for (int i = 0; i < 16; ++i) {
        const int row = rbase + (i << 3);                  // 0..127
        const float4* p =
            (const float4*)(h + (size_t)(n0 + row) * K_NB * CH) + c4;
        float4 acc = make_float4(0.f, 0.f, 0.f, 0.f);
#pragma unroll
        for (int k = 0; k < K_NB; ++k) {
            float4 t = p[(size_t)k * (CH / 4)];
            acc.x += t.x; acc.y += t.y; acc.z += t.z; acc.w += t.w;
        }
        sA[row][c4 * 2]     = pack2bf(acc.x, acc.y);
        sA[row][c4 * 2 + 1] = pack2bf(acc.z, acc.w);
    }
    __syncthreads();

    const int wave   = tid >> 5;
    const int lane   = tid & 31;
    const int lanelo = lane & 15;
    const int hi     = (lane >> 4) & 1;   // high half-wave

    v8f acc[8] = {};   // 8 N-tiles of 16 cols -> full 128-wide output

#pragma unroll
    for (int s = 0; s < 4; ++s) {         // K = 128 in 4 steps of 32
        // A fragment: 16x32 bf16, ISA layout (lanes 16-31 take K+8 / K+24)
        union { uint32_t u[8]; v16bf v; } af;
#pragma unroll
        for (int p = 0; p < 8; ++p) {
            const int half = p >> 2, jj = p & 3;
            const int k = 32 * s + (half ? 16 : 0) + (hi ? 8 : 0) + 2 * jj;
            af.u[p] = sA[wave * 16 + lanelo][k >> 1];
        }
#pragma unroll
        for (int t = 0; t < 8; ++t) {
            // B fragment: 32x16 bf16; lane=N col, hi lanes hold K+8 / K+24 (same as A)
            union { uint32_t u[8]; v16bf v; } bf;
#pragma unroll
            for (int p = 0; p < 8; ++p)
                bf.u[p] = sW[16 * t + lanelo][16 * s + (hi ? 4 : 0) + (p & 3) + ((p >> 2) << 3)];
            acc[t] = __builtin_amdgcn_wmma_f32_16x16x32_bf16(
                false, af.v, false, bf.v, (short)0, acc[t], false, false);
        }
    }

    // --- epilogue: +bo, store pre-BN, deterministic channel partials ---
    // D layout: lane -> col N = 16t+lanelo; VGPR r -> row M = 8*hi + r
#pragma unroll
    for (int t = 0; t < 8; ++t) {
        const int j = 16 * t + lanelo;
        const float b = bo[j];
        float s8 = 0.f, q8 = 0.f;
#pragma unroll
        for (int r = 0; r < 8; ++r) {
            const float v = acc[t][r] + b;
            const int row = n0 + wave * 16 + hi * 8 + r;
            outp[(size_t)row * CH + j] = v;
            s8 += v;
            q8 = fmaf(v, v, q8);
        }
        sRS[wave * 2 + hi][j] = s8;   // unique (wave,hi,t,lanelo) slot
        sRQ[wave * 2 + hi][j] = q8;
    }
    __syncthreads();
    if (tid < 128) {   // fixed-order reduction -> deterministic
        float s = 0.f, q = 0.f;
#pragma unroll
        for (int i = 0; i < 16; ++i) { s += sRS[i][tid]; q += sRQ[i][tid]; }
        part_sum[(size_t)blockIdx.x * CH + tid] = s;
        part_sq [(size_t)blockIdx.x * CH + tid] = q;
    }
}

// ---------------------------------------------------------------------------
// Kernel 3: fold NWG partials -> mean / inv-std per channel (fixed order)
// ---------------------------------------------------------------------------
__global__ void stats_kernel(const float* __restrict__ part_sum,
                             const float* __restrict__ part_sq,
                             float* __restrict__ mean_out,
                             float* __restrict__ invstd_out) {
    const int j = threadIdx.x;   // 128 threads
    float s = 0.f, q = 0.f;
    for (int w = 0; w < NWG; ++w) {
        s += part_sum[(size_t)w * CH + j];
        q += part_sq [(size_t)w * CH + j];
    }
    const float mean = s * (1.0f / (float)N_POINTS);
    const float var  = q * (1.0f / (float)N_POINTS) - mean * mean;
    mean_out[j]   = mean;
    invstd_out[j] = rsqrtf(var + BN_EPS);
}

// ---------------------------------------------------------------------------
// Kernel 4: in-place BatchNorm affine + ReLU over d_out (float4 vectorized)
// ---------------------------------------------------------------------------
__global__ __launch_bounds__(256) void bn_relu_kernel(
    float* __restrict__ out,
    const float* __restrict__ mean,
    const float* __restrict__ invstd,
    const float* __restrict__ gamma,
    const float* __restrict__ beta) {
    const int idx = blockIdx.x * blockDim.x + threadIdx.x;  // float4 index
    const int j4  = idx & (CH / 4 - 1);
    float4* o4 = (float4*)out;
    float4 v        = o4[idx];
    const float4 m  = ((const float4*)mean)[j4];
    const float4 is = ((const float4*)invstd)[j4];
    const float4 g  = ((const float4*)gamma)[j4];
    const float4 bt = ((const float4*)beta)[j4];
    v.x = fmaxf((v.x - m.x) * is.x * g.x + bt.x, 0.f);
    v.y = fmaxf((v.y - m.y) * is.y * g.y + bt.y, 0.f);
    v.z = fmaxf((v.z - m.z) * is.z * g.z + bt.z, 0.f);
    v.w = fmaxf((v.w - m.w) * is.w * g.w + bt.w, 0.f);
    o4[idx] = v;
}

// ---------------------------------------------------------------------------
// Inputs (setup_inputs order): 0:h 1:x 2:Wq 3:Wk 4:Wv 5:Wo 6:bo 7:gamma 8:beta
// x/Wq/Wk are dead (softmax over singleton axis == all-ones weights).
// ---------------------------------------------------------------------------
extern "C" void kernel_launch(void* const* d_in, const int* in_sizes, int n_in,
                              void* d_out, int out_size, void* d_ws, size_t ws_size,
                              hipStream_t stream) {
    (void)in_sizes; (void)n_in; (void)out_size; (void)ws_size;
    const float* h     = (const float*)d_in[0];
    const float* Wv    = (const float*)d_in[4];
    const float* Wo    = (const float*)d_in[5];
    const float* bo    = (const float*)d_in[6];
    const float* gamma = (const float*)d_in[7];
    const float* beta  = (const float*)d_in[8];
    float* out = (float*)d_out;

    char* ws = (char*)d_ws;
    uint32_t* WtP     = (uint32_t*)(ws);                          // 32 KB
    float*    part_sum= (float*)(ws + 32 * 1024);                 // 128 KB
    float*    part_sq = (float*)(ws + 32 * 1024 + 128 * 1024);    // 128 KB
    float*    meanArr = (float*)(ws + 32 * 1024 + 256 * 1024);    // 512 B
    float*    invstd  = meanArr + CH;                             // 512 B

    prep_w_kernel<<<32, 256, 0, stream>>>(Wv, Wo, WtP);
    main_kernel<<<NWG, 256, 0, stream>>>(h, WtP, bo, out, part_sum, part_sq);
    stats_kernel<<<1, 128, 0, stream>>>(part_sum, part_sq, meanArr, invstd);
    bn_relu_kernel<<<(N_POINTS * CH / 4) / 256, 256, 0, stream>>>(
        out, meanArr, invstd, gamma, beta);
}
